// RNNC_15642270892455
// MI455X (gfx1250) — matrix operand order repaired and checked by reading, not compiled, not run on hardware
//
#include <hip/hip_runtime.h>

// ---------------------------------------------------------------------------
// LSTM (B=512,T=1024,Din=64,H=256) + fused MLP(256->256->64) for gfx1250.
// Strategy: 32 workgroups x 16 batch rows; recurrence runs fully on-chip
// (h in LDS bf16, c in VGPRs), gates via v_wmma_f32_16x16x32_bf16 with
// fused [h|x] @ [W_hh|W_ih]^T (K=320). MLP fused per step from LDS h'.
// ---------------------------------------------------------------------------

typedef __attribute__((ext_vector_type(16))) __bf16 v16bf;
typedef __attribute__((ext_vector_type(8)))  float  v8f;

#define B_    512
#define T_    1024
#define DIN   64
#define H_    256
#define G4    1024      // 4*H
#define KT    320       // H + DIN  (fused K)
#define MLPH  256
#define DOUT  64

#define BM    16        // batch rows per workgroup
#define NW    16        // waves per workgroup
#define NTHR  (NW * 32)
#define SA    328       // LDS stride (bf16) for [h|x] buffer: 320 + 8 pad
#define SY    264       // LDS stride (bf16) for y1 buffer: 256 + 8 pad

// ws layout (bytes)
#define WS_WG   0            // combined gate weights bf16 [1024 x 320]
#define WS_W1   655360       // W1 bf16 [256 x 256]
#define WS_W2   786432       // W2 bf16 [64 x 256]
#define WS_BIAS 819200       // b_ih + b_hh, f32 [1024]

union Frag16 { v16bf v; uint4 u[2]; };

static __device__ inline __bf16 f2bf(float f) {
    union { float f; unsigned u; } a; a.f = f;
    unsigned u = a.u + 0x7FFFu + ((a.u >> 16) & 1u);   // round-to-nearest-even
    union { unsigned short s; __bf16 b; } o; o.s = (unsigned short)(u >> 16);
    return o.b;
}

static __device__ inline float sigm(float x) {
    return 1.0f / (1.0f + __expf(-x));
}
static __device__ inline float tanh_fast(float x) {
    float e = __expf(-2.0f * fabsf(x));
    float r = (1.0f - e) / (1.0f + e);
    return x >= 0.0f ? r : -r;
}

static __device__ inline v8f wmma_bf16(v16bf a, v16bf b, v8f c) {
    return __builtin_amdgcn_wmma_f32_16x16x32_bf16(
        false, a, false, b, (short)0, c, false, false);
}

// A fragment (16x32 bf16) from LDS, row-major with stride `stride` (bf16 elems)
// lane<16: rows 0..15, K = kk*32 + {0..7, 16..23}; lane>=16: K = +8
static __device__ inline v16bf lds_fragA(const __bf16* base, int stride, int lane, int kk) {
    int row  = lane & 15;
    int koff = kk * 32 + ((lane >> 4) << 3);
    Frag16 f;
    f.u[0] = *(const uint4*)(base + row * stride + koff);
    f.u[1] = *(const uint4*)(base + row * stride + koff + 16);
    return f.v;
}

// B fragment (32x16 bf16) from global row-major weights W[N x S]:
// lane holds column n = nt*16 + (lane&15), K = kk*32 + 8*(lane>>4) + {0..7, 16..23}
static __device__ inline v16bf glb_fragB(const __bf16* W, int S, int nt, int lane, int kk) {
    int col = nt * 16 + (lane & 15);
    int kb  = kk * 32 + ((lane >> 4) << 3);
    const uint4* p = (const uint4*)(W + col * S + kb);
    Frag16 f;
    f.u[0] = p[0];
    f.u[1] = p[2];
    return f.v;
}

// ---------------------------------------------------------------------------
// Prelude: build bf16 weight copies + fused bias in workspace.
// ---------------------------------------------------------------------------
__global__ void lstm_prep(const float* __restrict__ Wih, const float* __restrict__ Whh,
                          const float* __restrict__ bih, const float* __restrict__ bhh,
                          const float* __restrict__ W1,  const float* __restrict__ W2,
                          __bf16* __restrict__ Wg, __bf16* __restrict__ W1b,
                          __bf16* __restrict__ W2b, float* __restrict__ bias) {
    int i = blockIdx.x * blockDim.x + threadIdx.x;
    if (i < G4 * KT) {                      // fused [W_hh | W_ih] rows of length 320
        int n = i / KT, k = i - n * KT;
        float v = (k < H_) ? Whh[n * H_ + k] : Wih[n * DIN + (k - H_)];
        Wg[i] = f2bf(v);
        return;
    }
    int j = i - G4 * KT;
    if (j < MLPH * H_) { W1b[j] = f2bf(W1[j]); return; }
    int k2 = j - MLPH * H_;
    if (k2 < DOUT * MLPH) { W2b[k2] = f2bf(W2[k2]); return; }
    int b = k2 - DOUT * MLPH;
    if (b < G4) bias[b] = bih[b] + bhh[b];
}

// ---------------------------------------------------------------------------
// Main persistent-recurrence kernel: one workgroup = 16 batch rows, T steps.
// ---------------------------------------------------------------------------
__global__ __launch_bounds__(NTHR) void lstm_mlp(
    const float* __restrict__ x,   const float* __restrict__ mask,
    const float* __restrict__ h0,  const float* __restrict__ c0,
    const float* __restrict__ b1,  const float* __restrict__ b2,
    const __bf16* __restrict__ Wg, const __bf16* __restrict__ W1b,
    const __bf16* __restrict__ W2b,const float* __restrict__ bias,
    float* __restrict__ out) {

    __shared__ __bf16 hA[BM * SA];   // [h (0..255) | x_t (256..319)] per row, bf16
    __shared__ __bf16 y1[BM * SY];   // relu(h @ W1^T + b1), bf16
    __shared__ float  maskS[BM];

    const int tid  = threadIdx.x;
    const int lane = tid & 31;
    const int wv   = tid >> 5;           // wave id: owns hidden cols [16*wv, 16*wv+16)
    const int b0   = blockIdx.x * BM;
    const int colw = 16 * wv + (lane & 15);
    const int rhi  = (lane >> 4) << 3;   // 0 or 8: row offset inside C/D tile

    // --- init: h0 -> LDS (bf16), c0 -> registers ---
    for (int idx = tid; idx < BM * H_; idx += NTHR) {
        int r = idx >> 8, c = idx & (H_ - 1);
        hA[r * SA + c] = f2bf(h0[(b0 + r) * H_ + c]);
    }
    float creg[8];
#pragma unroll
    for (int r = 0; r < 8; ++r)
        creg[r] = c0[(size_t)(b0 + r + rhi) * H_ + colw];

    // per-wave constant biases (column-indexed, same for every row)
    const float bI = bias[0 * H_ + colw];
    const float bF = bias[1 * H_ + colw];
    const float bG = bias[2 * H_ + colw];
    const float bO = bias[3 * H_ + colw];
    const float b1v = b1[colw];
    const float b2v = (wv < 4) ? b2[colw] : 0.0f;

    __syncthreads();

    for (int t = 0; t < T_; ++t) {
        // ---- stage x_t (bf16) and mask into LDS ----
        for (int idx = tid; idx < BM * DIN; idx += NTHR) {
            int r = idx >> 6, c = idx & (DIN - 1);
            hA[r * SA + H_ + c] = f2bf(x[((size_t)(b0 + r) * T_ + t) * DIN + c]);
        }
        if (tid < BM) maskS[tid] = mask[(size_t)(b0 + tid) * T_ + t];
        __syncthreads();

        // ---- gates = [h|x] @ [W_hh|W_ih]^T ; wave wv does tiles i,f,g,o ----
        v8f aI = {0,0,0,0,0,0,0,0}, aF = aI, aG = aI, aO = aI;
#pragma unroll 2
        for (int kk = 0; kk < KT / 32; ++kk) {
            v16bf a  = lds_fragA(hA, SA, lane, kk);           // shared across 4 gates
            v16bf wi = glb_fragB(Wg, KT,  0 + wv, lane, kk);
            v16bf wf = glb_fragB(Wg, KT, 16 + wv, lane, kk);
            v16bf wg = glb_fragB(Wg, KT, 32 + wv, lane, kk);
            v16bf wo = glb_fragB(Wg, KT, 48 + wv, lane, kk);
            aI = wmma_bf16(a, wi, aI);
            aF = wmma_bf16(a, wf, aF);
            aG = wmma_bf16(a, wg, aG);
            aO = wmma_bf16(a, wo, aO);
        }
        __syncthreads();   // all waves done reading h before it is overwritten

        // ---- cell update (registers only), write h' back to LDS ----
        float hn[8];
#pragma unroll
        for (int r = 0; r < 8; ++r) {
            float iv = sigm(aI[r] + bI);
            float fv = sigm(aF[r] + bF);
            float gv = tanh_fast(aG[r] + bG);
            float ov = sigm(aO[r] + bO);
            float cv = fv * creg[r] + iv * gv;
            creg[r]  = cv;
            hn[r]    = ov * tanh_fast(cv);
        }
#pragma unroll
        for (int r = 0; r < 8; ++r)
            hA[(r + rhi) * SA + colw] = f2bf(hn[r]);
        __syncthreads();

        // ---- MLP layer 1: y1 = relu(h' @ W1^T + b1), wave wv -> cols [16wv,16wv+16) ----
        v8f a1 = {0,0,0,0,0,0,0,0};
#pragma unroll
        for (int kk = 0; kk < H_ / 32; ++kk) {
            v16bf a  = lds_fragA(hA, SA, lane, kk);
            v16bf w1 = glb_fragB(W1b, H_, wv, lane, kk);
            a1 = wmma_bf16(a, w1, a1);
        }
#pragma unroll
        for (int r = 0; r < 8; ++r) {
            float v = a1[r] + b1v;
            v = v > 0.0f ? v : 0.0f;
            y1[(r + rhi) * SY + colw] = f2bf(v);
        }
        __syncthreads();

        // ---- MLP layer 2 (waves 0..3): y = (y1 @ W2^T + b2) * mask ----
        if (wv < 4) {
            v8f a2 = {0,0,0,0,0,0,0,0};
#pragma unroll
            for (int kk = 0; kk < H_ / 32; ++kk) {
                v16bf a  = lds_fragA(y1, SY, lane, kk);
                v16bf w2 = glb_fragB(W2b, H_, wv, lane, kk);
                a2 = wmma_bf16(a, w2, a2);
            }
#pragma unroll
            for (int r = 0; r < 8; ++r) {
                int row = r + rhi;
                float v = (a2[r] + b2v) * maskS[row];
                out[((size_t)(b0 + row) * T_ + t) * DOUT + colw] = v;
            }
        }
        __syncthreads();   // protect maskS / y1 for next iteration
    }
}

extern "C" void kernel_launch(void* const* d_in, const int* in_sizes, int n_in,
                              void* d_out, int out_size, void* d_ws, size_t ws_size,
                              hipStream_t stream) {
    const float* x    = (const float*)d_in[0];
    const float* mask = (const float*)d_in[1];
    const float* h0   = (const float*)d_in[2];
    const float* c0   = (const float*)d_in[3];
    const float* Wih  = (const float*)d_in[4];
    const float* Whh  = (const float*)d_in[5];
    const float* bih  = (const float*)d_in[6];
    const float* bhh  = (const float*)d_in[7];
    const float* W1   = (const float*)d_in[8];
    const float* b1   = (const float*)d_in[9];
    const float* W2   = (const float*)d_in[10];
    const float* b2   = (const float*)d_in[11];
    float* out = (float*)d_out;

    char* ws = (char*)d_ws;
    __bf16* Wg   = (__bf16*)(ws + WS_WG);
    __bf16* W1b  = (__bf16*)(ws + WS_W1);
    __bf16* W2b  = (__bf16*)(ws + WS_W2);
    float*  bias = (float*)(ws + WS_BIAS);

    // prelude: bf16 weight packing + fused bias (410,624 work items)
    const int prep_total = G4 * KT + MLPH * H_ + DOUT * MLPH + G4;
    lstm_prep<<<(prep_total + 255) / 256, 256, 0, stream>>>(
        Wih, Whh, bih, bhh, W1, W2, Wg, W1b, W2b, bias);

    // persistent recurrence: 32 workgroups x 512 threads (16 waves)
    lstm_mlp<<<B_ / BM, NTHR, 0, stream>>>(
        x, mask, h0, c0, b1, b2, Wg, W1b, W2b, bias, out);
}
